// DSCAN_54219667145286
// MI455X (gfx1250) — matrix operand order repaired and checked by reading, not compiled, not run on hardware
//
#include <hip/hip_runtime.h>
#include <hip/hip_bf16.h>

typedef __attribute__((ext_vector_type(16))) _Float16 v16h;
typedef __attribute__((ext_vector_type(8)))  _Float16 v8h;
typedef __attribute__((ext_vector_type(4)))  _Float16 v4h;
typedef __attribute__((ext_vector_type(8)))  float    v8f;

#define TM 128
#define TN 64
#define TK 32
#define LDA_S 40   // halves per A-tile row (80B; all chunk offsets 8B/16B aligned)
#define LDB_S 40   // halves per W-tile (N-major) row

// ---------------------------------------------------------------------------
// Weight pre-pass: W (K x N, fp32) -> Wt (N x K, f16). Runs once per weight.
// ---------------------------------------------------------------------------
__global__ void dscan_w_transpose_f16(const float* __restrict__ W,
                                      _Float16* __restrict__ Wt,
                                      int K, int N, int total)
{
    int idx = blockIdx.x * blockDim.x + threadIdx.x;
    if (idx >= total) return;
    int n = idx / K, k = idx - n * K;
    Wt[idx] = (_Float16)W[(size_t)k * N + n];
}

// ---------------------------------------------------------------------------
// Generic GEMM: C[M,N] = A[M,K] @ W[K,N] (+ bias); W pre-transposed (N x K f16).
// A fp32 or f16, C fp32 or f16, f32 WMMA accumulate. Double-buffered LDS:
// next-tile global loads issue before WMMAs, commit after -> latency hidden.
// M % 128 == 0, K % 32 == 0. N edge-guarded.
// ---------------------------------------------------------------------------
template<typename TA, typename TC>
__global__ __launch_bounds__(256)
void dscan_gemm_wmma(const TA* __restrict__ A, const _Float16* __restrict__ Wt,
                     const float* __restrict__ bias, TC* __restrict__ C,
                     int M, int N, int K)
{
    __shared__ _Float16 As[2][TM * LDA_S];   // 2 x 10240 B
    __shared__ _Float16 Ws[2][TN * LDB_S];   // 2 x  5120 B

    const int tid   = threadIdx.x;
    const int lane  = tid & 31;
    const int wid   = tid >> 5;
    const int wm    = (wid & 3) * 32;
    const int wn    = (wid >> 2) * 32;
    const int bm    = blockIdx.y * TM;
    const int bn    = blockIdx.x * TN;
    const int lhalf = (lane < 16) ? 0 : 1;
    const int lmod  = lane & 15;

    v8f acc[2][2] = {};

    float4 avf[4];   // staged A regs (fp32 path)
    v8h    avh[2];   // staged A regs (f16 path)
    v8h    wreg;     // staged W regs

    auto loadA = [&](int k0) {
        if constexpr (sizeof(TA) == 4) {
            #pragma unroll
            for (int i = 0; i < 4; ++i) {
                int idx = tid + i * 256;            // float4 index, 1024 total
                int m = idx >> 3, c = idx & 7;
                avf[i] = *(const float4*)(A + (size_t)(bm + m) * K + k0 + c * 4);
            }
        } else {
            #pragma unroll
            for (int i = 0; i < 2; ++i) {
                int idx = tid + i * 256;            // 8-half chunk, 512 total
                int m = idx >> 2, c = idx & 3;
                avh[i] = *(const v8h*)(A + (size_t)(bm + m) * K + k0 + c * 8);
            }
        }
    };
    auto loadW = [&](int k0) {
        int n = tid >> 2, c = tid & 3;
        wreg = v8h{};
        if (bn + n < N)
            wreg = *(const v8h*)(Wt + (size_t)(bn + n) * K + k0 + c * 8);
    };
    auto commit = [&](int buf) {
        if constexpr (sizeof(TA) == 4) {
            #pragma unroll
            for (int i = 0; i < 4; ++i) {
                int idx = tid + i * 256;
                int m = idx >> 3, c = idx & 7;
                v4h hv;
                hv[0] = (_Float16)avf[i].x; hv[1] = (_Float16)avf[i].y;
                hv[2] = (_Float16)avf[i].z; hv[3] = (_Float16)avf[i].w;
                *(v4h*)(&As[buf][m * LDA_S + c * 4]) = hv;
            }
        } else {
            #pragma unroll
            for (int i = 0; i < 2; ++i) {
                int idx = tid + i * 256;
                int m = idx >> 2, c = idx & 3;
                *(v8h*)(&As[buf][m * LDA_S + c * 8]) = avh[i];
            }
        }
        int n = tid >> 2, c = tid & 3;
        *(v8h*)(&Ws[buf][n * LDB_S + c * 8]) = wreg;
    };

    // prologue: stage tile 0
    loadA(0); loadW(0);
    commit(0);
    __syncthreads();

    const int nk = K / TK;
    for (int it = 0; it < nk; ++it) {
        const int cur = it & 1;
        const bool has_next = (it + 1 < nk);

        // issue next-tile global loads first (in flight during WMMAs)
        if (has_next) {
            loadA((it + 1) * TK);
            loadW((it + 1) * TK);
            if (it + 2 < nk && tid < TM)
                __builtin_prefetch(A + (size_t)(bm + tid) * K + (it + 2) * TK, 0, 3);
        }

        // fragments from current buffer
        v16h afrag[2], bfrag[2];
        #pragma unroll
        for (int i = 0; i < 2; ++i) {
            const _Float16* rp = &As[cur][(wm + i * 16 + lmod) * LDA_S];
            v8h lo = *(const v8h*)(rp + lhalf * 8);
            v8h hi = *(const v8h*)(rp + 16 + lhalf * 8);
            #pragma unroll
            for (int j = 0; j < 8; ++j) { afrag[i][j] = lo[j]; afrag[i][j + 8] = hi[j]; }
        }
        #pragma unroll
        for (int i = 0; i < 2; ++i) {
            const _Float16* cp = &Ws[cur][(wn + i * 16 + lmod) * LDB_S + lhalf * 8];
            v8h lo = *(const v8h*)(cp);
            v8h hi = *(const v8h*)(cp + 16);
            #pragma unroll
            for (int j = 0; j < 8; ++j) { bfrag[i][j] = lo[j]; bfrag[i][j + 8] = hi[j]; }
        }
        #pragma unroll
        for (int i = 0; i < 2; ++i)
            #pragma unroll
            for (int j = 0; j < 2; ++j)
                acc[i][j] = __builtin_amdgcn_wmma_f32_16x16x32_f16(
                    false, afrag[i], false, bfrag[j], (short)0, acc[i][j], false, false);

        // commit next tile to alternate buffer, single barrier per iteration
        if (has_next) {
            commit(cur ^ 1);
            __syncthreads();
        }
    }

    #pragma unroll
    for (int i = 0; i < 2; ++i)
        #pragma unroll
        for (int j = 0; j < 2; ++j) {
            int col = bn + wn + j * 16 + lmod;
            if (col < N) {
                float bv = bias ? bias[col] : 0.f;
                #pragma unroll
                for (int r = 0; r < 8; ++r) {
                    int row = bm + wm + i * 16 + r + (lhalf ? 8 : 0);
                    float val = acc[i][j][r] + bv;
                    C[(size_t)row * N + col] = (TC)val;
                }
            }
        }
}

// ---------------------------------------------------------------------------
// Depthwise causal conv1d (kernel 4) + SiLU over (B,8,384) channels.
// ---------------------------------------------------------------------------
__global__ void dscan_conv_silu(const float* __restrict__ zx,      // (B*8,648)
                                const float* __restrict__ convw,   // (4,384)
                                const float* __restrict__ convb,   // (384,)
                                float* __restrict__ out, int total)
{
    int idx = blockIdx.x * blockDim.x + threadIdx.x;
    if (idx >= total) return;
    int c = idx % 384;
    int l = (idx / 384) & 7;
    int b = idx / (384 * 8);
    float acc = convb[c];
    #pragma unroll
    for (int k = 0; k < 4; ++k) {
        int t = l + k - 3;
        if (t >= 0) acc += zx[(size_t)(b * 8 + t) * 648 + 256 + c] * convw[k * 384 + c];
    }
    out[idx] = acc / (1.f + __expf(-acc));   // SiLU
}

// ---------------------------------------------------------------------------
// Selective scan: 1 wave per head, lane = P index, state h[64] in VGPRs.
// ---------------------------------------------------------------------------
__global__ __launch_bounds__(256)
void dscan_mamba_scan(const float* __restrict__ zx,    // (B*8,648): dt at [640+h]
                      const float* __restrict__ xbc,   // (B*8,384): x|B|C
                      const float* __restrict__ dtb,
                      const float* __restrict__ Alog,
                      const float* __restrict__ Dp,
                      float* __restrict__ y)           // (B*8,256)
{
    int b    = blockIdx.x;
    int head = threadIdx.x >> 5;
    int p    = threadIdx.x & 31;
    __shared__ float sB[64], sC[64];

    float h[64];
    #pragma unroll
    for (int n = 0; n < 64; ++n) h[n] = 0.f;

    float Ah   = -__expf(Alog[head]);
    float dtbh = dtb[head];
    float Dh   = Dp[head];

    for (int t = 0; t < 8; ++t) {
        const float* row = xbc + (size_t)(b * 8 + t) * 384;
        if (threadIdx.x < 64)        sB[threadIdx.x]      = row[256 + threadIdx.x];
        else if (threadIdx.x < 128)  sC[threadIdx.x - 64] = row[320 + threadIdx.x - 64];
        __syncthreads();

        float dtraw = zx[(size_t)(b * 8 + t) * 648 + 640 + head] + dtbh;
        float dt = (dtraw > 20.f) ? dtraw : __logf(1.f + __expf(dtraw));
        float dA = __expf(dt * Ah);
        float xv = row[head * 32 + p];
        float dtx = dt * xv;
        float yv = 0.f;
        #pragma unroll
        for (int n = 0; n < 64; ++n) {
            h[n] = h[n] * dA + sB[n] * dtx;
            yv  += h[n] * sC[n];
        }
        y[(size_t)(b * 8 + t) * 256 + head * 32 + p] = yv + Dh * xv;
        __syncthreads();
    }
}

// ---------------------------------------------------------------------------
// out16 = f16( rmsnorm(y * silu(z)) * normw ), 256 channels per row.
// ---------------------------------------------------------------------------
__global__ __launch_bounds__(256)
void dscan_gate_rmsnorm(const float* __restrict__ y, const float* __restrict__ zx,
                        const float* __restrict__ normw, _Float16* __restrict__ out16)
{
    int row = blockIdx.x;
    int c   = threadIdx.x;
    float z = zx[(size_t)row * 648 + c];
    float v = y[(size_t)row * 256 + c] * (z / (1.f + __expf(-z)));
    __shared__ float red[256];
    red[c] = v * v;
    __syncthreads();
    for (int s = 128; s > 0; s >>= 1) {
        if (c < s) red[c] += red[c + s];
        __syncthreads();
    }
    float scale = rsqrtf(red[0] * (1.f / 256.f) + 1e-5f);
    out16[(size_t)row * 256 + c] = (_Float16)(v * scale * normw[c]);
}

// ---------------------------------------------------------------------------
// Tiny attention core per batch: S = QK^T/sqrt(128), softmax, ctx16 = S V.
// ---------------------------------------------------------------------------
__global__ __launch_bounds__(128)
void dscan_attn_mid(const float* __restrict__ Q, const float* __restrict__ Kt,
                    const float* __restrict__ V, _Float16* __restrict__ O16)
{
    int b = blockIdx.x;
    int tid = threadIdx.x;
    __shared__ float sQ[8][128], sK[8][128], sA[8][8];

    for (int i = tid; i < 8 * 128; i += 128) {
        sQ[i >> 7][i & 127] = Q [(size_t)(b * 8 + (i >> 7)) * 128 + (i & 127)];
        sK[i >> 7][i & 127] = Kt[(size_t)(b * 8 + (i >> 7)) * 128 + (i & 127)];
    }
    __syncthreads();
    if (tid < 64) {
        int q = tid >> 3, kk = tid & 7;
        float s = 0.f;
        for (int d = 0; d < 128; ++d) s += sQ[q][d] * sK[kk][d];
        sA[q][kk] = s * 0.08838834764831845f;
    }
    __syncthreads();
    if (tid < 8) {
        float mx = sA[tid][0];
        #pragma unroll
        for (int k = 1; k < 8; ++k) mx = fmaxf(mx, sA[tid][k]);
        float e[8], sum = 0.f;
        #pragma unroll
        for (int k = 0; k < 8; ++k) { e[k] = __expf(sA[tid][k] - mx); sum += e[k]; }
        float inv = 1.f / sum;
        #pragma unroll
        for (int k = 0; k < 8; ++k) sA[tid][k] = e[k] * inv;
    }
    __syncthreads();
    for (int q = 0; q < 8; ++q) {
        float o = 0.f;
        #pragma unroll
        for (int k = 0; k < 8; ++k) o += sA[q][k] * V[(size_t)(b * 8 + k) * 128 + tid];
        O16[(size_t)(b * 8 + q) * 128 + tid] = (_Float16)o;
    }
}

// ---------------------------------------------------------------------------
// concat16([st_a16, exp_a16], -1).reshape(B, 2048)
// ---------------------------------------------------------------------------
__global__ void dscan_concat(const _Float16* __restrict__ st_a,
                             const _Float16* __restrict__ exp_a,
                             _Float16* __restrict__ out, int total)
{
    int idx = blockIdx.x * blockDim.x + threadIdx.x;
    if (idx >= total) return;
    int b = idx >> 11, j = idx & 2047;
    int l = j >> 8, j2 = j & 255;
    out[idx] = (j2 < 128) ? st_a[(size_t)(b * 8 + l) * 128 + j2]
                          : exp_a[(size_t)(b * 8 + l) * 128 + (j2 - 128)];
}

// ---------------------------------------------------------------------------
extern "C" void kernel_launch(void* const* d_in, const int* in_sizes, int n_in,
                              void* d_out, int out_size, void* d_ws, size_t ws_size,
                              hipStream_t stream)
{
    const float* st_feature  = (const float*)d_in[0];
    const float* exp_feature = (const float*)d_in[1];
    const float* mam[2][8];
    for (int i = 0; i < 8; ++i) { mam[0][i] = (const float*)d_in[2 + i];
                                  mam[1][i] = (const float*)d_in[10 + i]; }
    const float* cx[8]; const float* cs[8];
    for (int i = 0; i < 8; ++i) { cx[i] = (const float*)d_in[18 + i];
                                  cs[i] = (const float*)d_in[26 + i]; }
    const float* mlp_W = (const float*)d_in[34];
    const float* mlp_b = (const float*)d_in[35];

    // ---- fp32 workspace ----
    float* wsf = (float*)d_ws;
    size_t off = 0;
    float* zx   = wsf + off; off += (size_t)16384 * 648;  // in-proj out; later Q|K|V
    float* ybuf = wsf + off; off += (size_t)16384 * 256;  // scan output (pre-gate)
    float* conv = wsf + off; off += (size_t)16384 * 384;  // conv+SiLU out
    float* Qb = zx;
    float* Kb = zx + (size_t)16384 * 128;
    float* Vb = zx + (size_t)2 * 16384 * 128;

    // ---- f16 workspace ----
    _Float16* wsh = (_Float16*)(wsf + off);
    size_t h = 0;
    _Float16* h16    = wsh + h; h += (size_t)16384 * 256; // gated-y / ctx16 / concat16
    _Float16* exf16  = wsh + h; h += (size_t)16384 * 128;
    _Float16* stf16  = wsh + h; h += (size_t)16384 * 128;
    _Float16* expa16 = wsh + h; h += (size_t)16384 * 128;
    _Float16* sta16  = wsh + h; h += (size_t)16384 * 128;
    // transposed f16 weights
    _Float16* wt[2][2]; _Float16* wtcx[4]; _Float16* wtcs[4]; _Float16* wtmlp;
    for (int m = 0; m < 2; ++m) {
        wt[m][0] = wsh + h; h += (size_t)128 * 648;   // Win^T
        wt[m][1] = wsh + h; h += (size_t)256 * 128;   // Wout^T
    }
    for (int i = 0; i < 4; ++i) { wtcx[i] = wsh + h; h += (size_t)128 * 128; }
    for (int i = 0; i < 4; ++i) { wtcs[i] = wsh + h; h += (size_t)128 * 128; }
    wtmlp = wsh + h; h += (size_t)2048 * 1024;

    auto wtrans = [&](const float* W, _Float16* Wt, int K, int N) {
        int total = K * N;
        dscan_w_transpose_f16<<<(total + 255) / 256, 256, 0, stream>>>(W, Wt, K, N, total);
    };
    wtrans(mam[0][0], wt[0][0], 128, 648);  wtrans(mam[0][7], wt[0][1], 256, 128);
    wtrans(mam[1][0], wt[1][0], 128, 648);  wtrans(mam[1][7], wt[1][1], 256, 128);
    for (int i = 0; i < 4; ++i) { wtrans(cx[2 * i], wtcx[i], 128, 128);
                                  wtrans(cs[2 * i], wtcs[i], 128, 128); }
    wtrans(mlp_W, wtmlp, 2048, 1024);

    auto gemm_ff = [&](const float* A, const _Float16* Wt, const float* bias, float* C,
                       int M, int N, int K) {
        dim3 g((N + TN - 1) / TN, M / TM);
        dscan_gemm_wmma<float, float><<<g, 256, 0, stream>>>(A, Wt, bias, C, M, N, K);
    };
    auto gemm_hf = [&](const _Float16* A, const _Float16* Wt, const float* bias, float* C,
                       int M, int N, int K) {
        dim3 g((N + TN - 1) / TN, M / TM);
        dscan_gemm_wmma<_Float16, float><<<g, 256, 0, stream>>>(A, Wt, bias, C, M, N, K);
    };
    auto gemm_hh = [&](const _Float16* A, const _Float16* Wt, const float* bias, _Float16* C,
                       int M, int N, int K) {
        dim3 g((N + TN - 1) / TN, M / TM);
        dscan_gemm_wmma<_Float16, _Float16><<<g, 256, 0, stream>>>(A, Wt, bias, C, M, N, K);
    };

    auto run_mamba = [&](const float* u, const float* const* p, const _Float16* const* w,
                         _Float16* fout16) {
        gemm_ff(u, w[0], nullptr, zx, 16384, 648, 128);
        dscan_conv_silu<<<(16384 * 384 + 255) / 256, 256, 0, stream>>>(zx, p[1], p[2], conv, 16384 * 384);
        dscan_mamba_scan<<<2048, 256, 0, stream>>>(zx, conv, p[3], p[4], p[5], ybuf);
        dscan_gate_rmsnorm<<<16384, 256, 0, stream>>>(ybuf, zx, p[6], h16);
        gemm_hh(h16, w[1], nullptr, fout16, 16384, 128, 256);
    };

    auto run_attn = [&](const _Float16* xq, const _Float16* ctx, const float* const* wb,
                        _Float16* const* w, _Float16* out16) {
        gemm_hf(xq,  w[0], wb[1], Qb, 16384, 128, 128);
        gemm_hf(ctx, w[1], wb[3], Kb, 16384, 128, 128);
        gemm_hf(ctx, w[2], wb[5], Vb, 16384, 128, 128);
        dscan_attn_mid<<<2048, 128, 0, stream>>>(Qb, Kb, Vb, h16);
        gemm_hh(h16, w[3], wb[7], out16, 16384, 128, 128);
    };

    run_mamba(exp_feature, mam[1], wt[1], exf16);   // ex params -> exp_f
    run_mamba(st_feature,  mam[0], wt[0], stf16);   // st params -> st_f
    run_attn(exf16, stf16, cx, wtcx, expa16);       // exp_a: Q from exp_f, K/V from st_f
    run_attn(stf16, exf16, cs, wtcs, sta16);        // st_a : Q from st_f, K/V from exp_f
    dscan_concat<<<(2048 * 2048 + 255) / 256, 256, 0, stream>>>(sta16, expa16, h16, 2048 * 2048);
    gemm_hf(h16, wtmlp, mlp_b, (float*)d_out, 2048, 1024, 2048);
}
